// PairwiseScore_910533066865
// MI455X (gfx1250) — matrix operand order repaired and checked by reading, not compiled, not run on hardware
//
#include <hip/hip_runtime.h>

typedef __attribute__((ext_vector_type(16))) _Float16 v16h;
typedef __attribute__((ext_vector_type(8)))  _Float16 v8h;
typedef __attribute__((ext_vector_type(8)))  float    v8f;
typedef __attribute__((ext_vector_type(4)))  float    v4f;

#define GD    512
#define GIJ   1596
#define H1    150
#define HPAD  160
#define NT1   10     // N tiles of 16 (150 -> 160)
#define KT1   50     // K tiles of 32 (1596 -> 1600)
#define KT2   5      // K tiles of 32 (150 -> 160)
#define W1_HALVES (KT1*NT1*32*16)
#define W2_HALVES (KT2*NT1*32*16)
#define BSTEP_U2  1280   // 10 tiles * 512 halves = 10240 B = 1280 uint2 per K-step

// ---------------- weight pre-pack (fp32 -> f16, WMMA B-register layout) -------------
// B 32x16 (KxN): lane L holds 16 contiguous halves: n = nt*16 + (L&15),
// k = kt*32 + (L>>4)*8 + (h<8 ? h : h+8)
__global__ void pack_w1_kernel(const float* __restrict__ W, _Float16* __restrict__ out) {
  int tile = blockIdx.x;                 // kt*NT1 + nt
  int kt = tile / NT1, nt = tile % NT1;
  int lane = threadIdx.x & 31;
  int n  = nt * 16 + (lane & 15);
  int kb = kt * 32 + (lane >> 4) * 8;
  union { _Float16 h[16]; v16h v; } u;
#pragma unroll
  for (int t = 0; t < 16; ++t) {
    int k = kb + (t < 8 ? t : t + 8);
    float x = (k < GIJ && n < H1) ? W[k * H1 + n] : 0.0f;
    u.h[t] = (_Float16)x;
  }
  ((v16h*)out)[tile * 32 + lane] = u.v;
}

__global__ void pack_w2_kernel(const float* __restrict__ W, _Float16* __restrict__ out) {
  int tile = blockIdx.x;                 // kt*NT1 + nt, kt in [0,5)
  int kt = tile / NT1, nt = tile % NT1;
  int lane = threadIdx.x & 31;
  int n  = nt * 16 + (lane & 15);
  int kb = kt * 32 + (lane >> 4) * 8;
  union { _Float16 h[16]; v16h v; } u;
#pragma unroll
  for (int t = 0; t < 16; ++t) {
    int k = kb + (t < 8 ? t : t + 8);
    float x = (k < H1 && n < H1) ? W[k * H1 + n] : 0.0f;
    u.h[t] = (_Float16)x;
  }
  ((v16h*)out)[tile * 32 + lane] = u.v;
}

// ---------------- shared-memory overlay ----------------
union TailH1 {
  _Float16 tail[8][16][64];     // embedding tail (layer-1 K-tiles 48..49)
  _Float16 h1[8][16][HPAD];     // layer-1 activations (time-disjoint with tail)
};

// ---------------- fused pair-MLP kernel ----------------
__launch_bounds__(256)
__global__ void pair_mlp_kernel(
    const float* __restrict__ g,  const float* __restrict__ si,
    const int* __restrict__ i1,   const int* __restrict__ width,
    const int* __restrict__ genre_idx, const int* __restrict__ speaker_idx,
    const int* __restrict__ i_idx, const int* __restrict__ j_idx,
    const float* __restrict__ dist_emb, const float* __restrict__ genre_emb,
    const float* __restrict__ speaker_emb,
    const float* __restrict__ b1, const float* __restrict__ b2,
    const float* __restrict__ W3, const float* __restrict__ b3,
    const _Float16* __restrict__ packW1, const _Float16* __restrict__ packW2,
    float* __restrict__ out, int P)
{
  __shared__ __align__(16) _Float16 s_B[2][NT1 * 32 * 16];  // 2 x 10 KB, double buffer
  __shared__ __align__(16) TailH1 s_u;                      // 40 KB overlay

  const int tid  = threadIdx.x;
  const int lane = tid & 31;
  const int wave = tid >> 5;
  const int r  = lane & 15;     // pair row within tile (A rows / C columns)
  const int hs = lane >> 4;     // K-half selector for A layout
  const int rowbase = blockIdx.x * 128 + wave * 16;

  // ---- stage 60-feature embedding tail (+4 zero pad) into LDS, f16 ----
  {
    int rr  = lane >> 1;
    int seg = (lane & 1) * 32;
    int p = rowbase + rr; if (p >= P) p = P - 1;
    int i = i_idx[p], j = j_idx[p];
    int dist = i1[i] + width[i] - i1[j];
    int bin = (dist >= 1) + (dist >= 2) + (dist >= 3) + (dist >= 4) +
              (dist >= 8) + (dist >= 16) + (dist >= 32) + (dist >= 64);
    const float* de = dist_emb    + bin * 20;
    const float* ge = genre_emb   + genre_idx[p] * 20;
    const float* se = speaker_emb + speaker_idx[p] * 20;
#pragma unroll
    for (int t = 0; t < 32; ++t) {
      int kk = seg + t;
      float v = 0.0f;
      if      (kk < 20) v = de[kk];
      else if (kk < 40) v = ge[kk - 20];
      else if (kk < 60) v = se[kk - 40];
      s_u.tail[wave][rr][kk] = (_Float16)v;
    }
  }

  int p = rowbase + r; if (p >= P) p = P - 1;
  const float* gip = g + (size_t)i_idx[p] * GD;
  const float* gjp = g + (size_t)j_idx[p] * GD;

  const v8f vz = {0, 0, 0, 0, 0, 0, 0, 0};
  v8f acc[NT1];
#pragma unroll
  for (int n = 0; n < NT1; ++n) acc[n] = vz;

  union AU { _Float16 h[16]; v16h v; } A;
  union BU { _Float16 h[16]; v16h v; v8h q[2]; } bu;

  // ---------- B-pipeline state (cooperative LDS staging, no divergence) ----------
  const uint2* srcW1 = (const uint2*)packW1;
  uint2 bc[5];
  // ---------- A-pipeline state ----------
  v4f fa0{}, fa1{}, fa2{}, fa3{}, fb0{}, fb1{}, fb2{}, fb3{};

  auto loadAf = [&](int kt) {
    int c0 = (kt & 15) * 32 + hs * 8;
    int sel = kt >> 4;
    if (sel != 1) {               // need gi
      fa0 = *(const v4f*)(gip + c0);      fa1 = *(const v4f*)(gip + c0 + 4);
      fa2 = *(const v4f*)(gip + c0 + 16); fa3 = *(const v4f*)(gip + c0 + 20);
    }
    if (sel != 0) {               // need gj
      fb0 = *(const v4f*)(gjp + c0);      fb1 = *(const v4f*)(gjp + c0 + 4);
      fb2 = *(const v4f*)(gjp + c0 + 16); fb3 = *(const v4f*)(gjp + c0 + 20);
    }
  };
  auto loadBregs = [&](const uint2* src) {
#pragma unroll
    for (int c = 0; c < 5; ++c) bc[c] = src[tid + c * 256];
  };
  auto storeBlds = [&](int buf) {
    uint2* dst = (uint2*)s_B[buf];
#pragma unroll
    for (int c = 0; c < 5; ++c) dst[tid + c * 256] = bc[c];
  };
  auto wmma10 = [&](int buf) {
    const _Float16* Bc = s_B[buf];
#pragma unroll
    for (int nt = 0; nt < NT1; ++nt) {
      const _Float16* bp = Bc + (nt * 32 + lane) * 16;
      bu.q[0] = *(const v8h*)bp;
      bu.q[1] = *(const v8h*)(bp + 8);
      acc[nt] = __builtin_amdgcn_wmma_f32_16x16x32_f16(false, A.v, false, bu.v,
                                                       (short)0, acc[nt], false, false);
    }
  };

  // ---- pipeline prologue: B(kt=0) -> LDS buf0, B(kt=1) -> regs, A(kt=0) -> regs ----
  loadBregs(srcW1);
  storeBlds(0);
  loadBregs(srcW1 + BSTEP_U2);
  loadAf(0);
  __syncthreads();

  // ================= layer 1: 50 K-steps, 10 WMMAs each =================
  for (int kt = 0; kt < KT1; ++kt) {
    const int cur = kt & 1;
    if (kt + 1 < KT1) storeBlds(cur ^ 1);                       // regs hold kt+1
    if (kt + 2 < KT1) loadBregs(srcW1 + (size_t)(kt + 2) * BSTEP_U2);

    // build A(kt) from floats prefetched last iteration
    if (kt < 16) {
#pragma unroll
      for (int t = 0; t < 4; ++t) {
        A.h[t]      = (_Float16)fa0[t]; A.h[4 + t]  = (_Float16)fa1[t];
        A.h[8 + t]  = (_Float16)fa2[t]; A.h[12 + t] = (_Float16)fa3[t];
      }
    } else if (kt < 32) {
#pragma unroll
      for (int t = 0; t < 4; ++t) {
        A.h[t]      = (_Float16)fb0[t]; A.h[4 + t]  = (_Float16)fb1[t];
        A.h[8 + t]  = (_Float16)fb2[t]; A.h[12 + t] = (_Float16)fb3[t];
      }
    } else if (kt < 48) {
#pragma unroll
      for (int t = 0; t < 4; ++t) {
        A.h[t]      = (_Float16)(fa0[t] * fb0[t]); A.h[4 + t]  = (_Float16)(fa1[t] * fb1[t]);
        A.h[8 + t]  = (_Float16)(fa2[t] * fb2[t]); A.h[12 + t] = (_Float16)(fa3[t] * fb3[t]);
      }
    } else {
      int c0 = (kt - 48) * 32 + hs * 8;
      *(v8h*)(A.h)     = *(const v8h*)&s_u.tail[wave][r][c0];
      *(v8h*)(A.h + 8) = *(const v8h*)&s_u.tail[wave][r][c0 + 16];
    }
    if (kt + 1 < 48) loadAf(kt + 1);                            // prefetch A floats

    wmma10(cur);
    __syncthreads();
  }

  // ---- bias + relu, transpose via LDS (C layout -> row-major f16) ----
  // tail region is dead now (last read before final loop barrier), safe to overwrite h1
#pragma unroll
  for (int nt = 0; nt < NT1; ++nt) {
    int n = nt * 16 + r;
    float bb = (n < H1) ? b1[n] : 0.0f;
#pragma unroll
    for (int v = 0; v < 8; ++v) {
      int m = v + 8 * hs;
      float h = acc[nt][v] + bb;
      h = h > 0.0f ? h : 0.0f;
      s_u.h1[wave][m][n] = (_Float16)h;
    }
  }

  // ---- layer-2 B pipeline prologue (s_B free after last layer-1 barrier) ----
  const uint2* srcW2 = (const uint2*)packW2;
  loadBregs(srcW2);
  storeBlds(0);
  loadBregs(srcW2 + BSTEP_U2);
#pragma unroll
  for (int n = 0; n < NT1; ++n) acc[n] = vz;
  __syncthreads();

  // ================= layer 2: 5 K-steps, 10 WMMAs each =================
  for (int kt = 0; kt < KT2; ++kt) {
    const int cur = kt & 1;
    if (kt + 1 < KT2) storeBlds(cur ^ 1);
    if (kt + 2 < KT2) loadBregs(srcW2 + (size_t)(kt + 2) * BSTEP_U2);

    int c0 = kt * 32 + hs * 8;
    *(v8h*)(A.h)     = *(const v8h*)&s_u.h1[wave][r][c0];
    *(v8h*)(A.h + 8) = *(const v8h*)&s_u.h1[wave][r][c0 + 16];

    wmma10(cur);
    __syncthreads();
  }

  // ---- layer 3: relu(h2 + b2) . W3, reduced across 16 lanes per half ----
  float part[8];
#pragma unroll
  for (int v = 0; v < 8; ++v) part[v] = 0.0f;
#pragma unroll
  for (int nt = 0; nt < NT1; ++nt) {
    int n = nt * 16 + r;
    float bb = (n < H1) ? b2[n] : 0.0f;
    float w3 = (n < H1) ? W3[n] : 0.0f;
#pragma unroll
    for (int v = 0; v < 8; ++v) {
      float h = acc[nt][v] + bb;
      h = h > 0.0f ? h : 0.0f;
      part[v] += h * w3;
    }
  }
#pragma unroll
  for (int v = 0; v < 8; ++v) {
    float x = part[v];
    x += __shfl_xor(x, 1, 32);
    x += __shfl_xor(x, 2, 32);
    x += __shfl_xor(x, 4, 32);
    x += __shfl_xor(x, 8, 32);
    part[v] = x;
  }
  if (r == 0) {                 // lanes 0 (rows 0..7) and 16 (rows 8..15)
    float bb3 = b3[0];
#pragma unroll
    for (int v = 0; v < 8; ++v) {
      int m = v + 8 * hs;
      int pm = rowbase + m;
      if (pm < P)
        out[pm] = part[v] + bb3 + si[i_idx[pm]] + si[j_idx[pm]];
    }
  }
}

extern "C" void kernel_launch(void* const* d_in, const int* in_sizes, int n_in,
                              void* d_out, int out_size, void* d_ws, size_t ws_size,
                              hipStream_t stream) {
  const float* g           = (const float*)d_in[0];
  const float* si          = (const float*)d_in[1];
  const int*   i1          = (const int*)  d_in[2];
  const int*   width       = (const int*)  d_in[3];
  const int*   genre_idx   = (const int*)  d_in[4];
  const int*   speaker_idx = (const int*)  d_in[5];
  const int*   i_idx       = (const int*)  d_in[6];
  const int*   j_idx       = (const int*)  d_in[7];
  const float* dist_emb    = (const float*)d_in[8];
  const float* genre_emb   = (const float*)d_in[9];
  const float* speaker_emb = (const float*)d_in[10];
  const float* W1          = (const float*)d_in[11];
  const float* b1          = (const float*)d_in[12];
  const float* W2          = (const float*)d_in[13];
  const float* b2          = (const float*)d_in[14];
  const float* W3          = (const float*)d_in[15];
  const float* b3          = (const float*)d_in[16];
  const int P = in_sizes[4];

  _Float16* pw1 = (_Float16*)d_ws;
  _Float16* pw2 = pw1 + W1_HALVES;

  pack_w1_kernel<<<KT1 * NT1, 32, 0, stream>>>(W1, pw1);
  pack_w2_kernel<<<KT2 * NT1, 32, 0, stream>>>(W2, pw2);

  int blocks = (P + 127) / 128;
  pair_mlp_kernel<<<blocks, 256, 0, stream>>>(
      g, si, i1, width, genre_idx, speaker_idx, i_idx, j_idx,
      dist_emb, genre_emb, speaker_emb, b1, b2, W3, b3,
      pw1, pw2, (float*)d_out, P);
}
